// MultiScaleAttention_31001073942935
// MI455X (gfx1250) — matrix operand order repaired and checked by reading, not compiled, not run on hardware
//
#include <hip/hip_runtime.h>
#include <hip/hip_bf16.h>
#include <stdint.h>

// ---------------------------------------------------------------------------
// Problem constants (from reference)
// ---------------------------------------------------------------------------
#define DIMC      768
#define NHEADS    8
#define HD        96
#define BATCH     4
#define T0        8
#define H0        28
#define W0        28
#define NTOK      6273          // 8*28*28 + 1
#define M1        (BATCH*NTOK)  // 25092 rows for QKV GEMM
#define M1TILES   1569          // ceil(25092/16) -> padded to 25104
#define M1PAD     (M1TILES*16)  // 25104
#define NPOOL     1569          // 8*14*14 + 1 pooled tokens
#define NPPAD     1600          // padded to 32-key multiple (50 tiles of 32)
#define KTILES    50
#define QTILES    99            // ceil(1569/16)
#define M2        (BATCH*NPOOL) // 6276 rows for proj GEMM
#define M2TILES   393           // ceil(6276/16) -> padded to 6288
#define M2PAD     (M2TILES*16)
#define SCALE_F   0.10206207261596575f  // 96^-0.5

typedef __attribute__((ext_vector_type(16))) __bf16 v16bf;
typedef __attribute__((ext_vector_type(8)))  __bf16 v8bf;
typedef __attribute__((ext_vector_type(8)))  float  v8f;

// ---------------------------------------------------------------------------
// WMMA helpers (CDNA5 16x16x32 bf16, f32 accumulate)
// ---------------------------------------------------------------------------
__device__ __forceinline__ v8f wmma_bf16(v16bf a, v16bf b, v8f c) {
    return __builtin_amdgcn_wmma_f32_16x16x32_bf16(
        /*neg_a=*/false, a, /*neg_b=*/false, b,
        /*c_mod=*/(short)0, c, /*reuse_a=*/false, /*reuse_b=*/false);
}

// A-matrix 16x32 bf16 fragment (ISA 7.12.2 layout):
//   lane half (lane>>4) selects K offset 8; elements 0..7 -> K = half*8+0..7,
//   elements 8..15 -> K = 16+half*8+0..7. Row M = lane&15. `base` points at
//   A[tileM][tileK], row-major with leading dim `ld` (bf16 elements).
__device__ __forceinline__ v16bf load_fragA16(const __bf16* base, int ld) {
    int lane = threadIdx.x & 31;
    int half = lane >> 4, m = lane & 15;
    const __bf16* p = base + (long)m * ld + half * 8;
    v8bf lo = *(const v8bf*)(p);
    v8bf hi = *(const v8bf*)(p + 16);
    v16bf f;
#pragma unroll
    for (int i = 0; i < 8; ++i) { f[i] = lo[i]; f[i + 8] = hi[i]; }
    return f;
}

// B-matrix 32x16 bf16 fragment. Column N = lane&15; elements 0..7 -> K =
// half*8+0..7, elements 8..15 -> K = 16+half*8+0..7. `base` points at B-source
// row-major [N][K] (i.e. W[n][k]) at [tileN][tileK]; leading dim `ld`.
__device__ __forceinline__ v16bf load_fragB16(const __bf16* base, int ld) {
    int lane = threadIdx.x & 31;
    int half = lane >> 4, n = lane & 15;
    const __bf16* p = base + (long)n * ld + half * 8;
    v8bf lo = *(const v8bf*)(p);
    v8bf hi = *(const v8bf*)(p + 16);
    v16bf f;
#pragma unroll
    for (int i = 0; i < 8; ++i) { f[i] = lo[i]; f[i + 8] = hi[i]; }
    return f;
}

// ---------------------------------------------------------------------------
// Kernel 1: zero fill (u32 grid-stride) for padded staging buffers
// ---------------------------------------------------------------------------
__global__ void k_zero_u32(uint32_t* p, long n) {
    long i = (long)blockIdx.x * blockDim.x + threadIdx.x;
    long stride = (long)gridDim.x * blockDim.x;
    for (; i < n; i += stride) p[i] = 0u;
}

// ---------------------------------------------------------------------------
// Kernel 2: f32 -> bf16 convert of x and the 4 weight matrices
// ---------------------------------------------------------------------------
__global__ void k_convert(const float* __restrict__ x,
                          const float* __restrict__ wq, const float* __restrict__ wk,
                          const float* __restrict__ wv, const float* __restrict__ wp,
                          __bf16* __restrict__ xb, __bf16* __restrict__ wb) {
    const long XT = (long)M1 * DIMC;          // 19,270,656
    const long WT = (long)DIMC * DIMC;        // 589,824
    const long TOT = XT + 4 * WT;
    long i = (long)blockIdx.x * blockDim.x + threadIdx.x;
    long stride = (long)gridDim.x * blockDim.x;
    for (; i < TOT; i += stride) {
        if (i < XT) {
            xb[i] = (__bf16)x[i];
        } else {
            long j = i - XT;
            int mat = (int)(j / WT);
            long r = j % WT;
            const float* s = (mat == 0) ? wq : (mat == 1) ? wk : (mat == 2) ? wv : wp;
            wb[(long)mat * WT + r] = (__bf16)s[r];
        }
    }
}

// ---------------------------------------------------------------------------
// Kernel 3: QKV GEMM (bf16 WMMA, f32 acc). One wave per 16x16 output tile,
// A fragment of x reused across the 3 weight matrices. Output scattered into
// (B, NHEADS, NTOK, HD) bf16 layout.
// ---------------------------------------------------------------------------
__global__ void __launch_bounds__(256)
k_qkv_gemm(const __bf16* __restrict__ xb, const __bf16* __restrict__ wb,
           const float* __restrict__ bq, const float* __restrict__ bk,
           const float* __restrict__ bv,
           __bf16* __restrict__ qpre, __bf16* __restrict__ kpre,
           __bf16* __restrict__ vpre) {
    const int NT = DIMC / 16;                           // 48
    int wave = blockIdx.x * (blockDim.x >> 5) + (threadIdx.x >> 5);
    if (wave >= M1TILES * NT) return;
    int mt = wave / NT, nt = wave % NT;

    v8f acc[3] = {};
    const __bf16* abase = xb + (long)mt * 16 * DIMC;
    const long WT = (long)DIMC * DIMC;
#pragma unroll 4
    for (int kk = 0; kk < DIMC; kk += 32) {
        __builtin_prefetch(abase + kk + 64, 0, 1);
        v16bf a = load_fragA16(abase + kk, DIMC);
#pragma unroll
        for (int m3 = 0; m3 < 3; ++m3) {
            v16bf b = load_fragB16(wb + m3 * WT + (long)nt * 16 * DIMC + kk, DIMC);
            acc[m3] = wmma_bf16(a, b, acc[m3]);
        }
    }

    int lane = threadIdx.x & 31, half = lane >> 4, n = lane & 15;
    int col = nt * 16 + n;
    int h = col / HD, d = col % HD;
    const float* biases[3] = {bq, bk, bv};
    __bf16* outs[3] = {qpre, kpre, vpre};
#pragma unroll
    for (int m3 = 0; m3 < 3; ++m3) {
        float bias = biases[m3][col];
#pragma unroll
        for (int r = 0; r < 8; ++r) {
            int m = mt * 16 + r + 8 * half;
            if (m < M1) {
                int b_ = m / NTOK, tok = m % NTOK;
                outs[m3][(((long)(b_ * NHEADS + h)) * NTOK + tok) * HD + d] =
                    (__bf16)(acc[m3][r] + bias);
            }
        }
    }
}

// ---------------------------------------------------------------------------
// Kernel 4: depthwise 3x3x3 conv pool (stride 1,2,2, pad 1) + LayerNorm.
// One wave per (b*h, pooled-token); each lane owns 3 channels of HD=96.
// Writes pooled q (f32 for residual + bf16), pooled k (bf16), V^T (bf16).
// ---------------------------------------------------------------------------
__global__ void __launch_bounds__(128)
k_pool_ln(const __bf16* __restrict__ qpre, const __bf16* __restrict__ kpre,
          const __bf16* __restrict__ vpre,
          const float* __restrict__ pwq, const float* __restrict__ pwk,
          const float* __restrict__ pwv,
          const float* __restrict__ gq, const float* __restrict__ bgq,
          const float* __restrict__ gk, const float* __restrict__ bgk,
          const float* __restrict__ gv, const float* __restrict__ bgv,
          float* __restrict__ qpf, __bf16* __restrict__ qpb,
          __bf16* __restrict__ kpb, __bf16* __restrict__ vtb) {
    int wid = blockIdx.x * (blockDim.x >> 5) + (threadIdx.x >> 5);
    if (wid >= BATCH * NHEADS * NPOOL) return;
    int bh = wid / NPOOL, o = wid % NPOOL;
    int lane = threadIdx.x & 31;

    const __bf16* ins[3] = {qpre, kpre, vpre};
    const float* pws[3] = {pwq, pwk, pwv};
    const float* gs[3]  = {gq, gk, gv};
    const float* bs[3]  = {bgq, bgk, bgv};

    for (int mat = 0; mat < 3; ++mat) {
        const __bf16* in = ins[mat] + (long)bh * NTOK * HD;
        float acc[3];
        if (o == 0) {
#pragma unroll
            for (int j = 0; j < 3; ++j) acc[j] = (float)in[lane + 32 * j];
        } else {
            int oi = o - 1;
            int to = oi / 196, rest = oi % 196, ho = rest / 14, wo = rest % 14;
#pragma unroll
            for (int j = 0; j < 3; ++j) acc[j] = 0.f;
            for (int kt = 0; kt < 3; ++kt) {
                int ti = to + kt - 1;
                if ((unsigned)ti >= (unsigned)T0) continue;
                for (int kh = 0; kh < 3; ++kh) {
                    int hi = 2 * ho + kh - 1;
                    if ((unsigned)hi >= (unsigned)H0) continue;
                    for (int kw = 0; kw < 3; ++kw) {
                        int wi = 2 * wo + kw - 1;
                        if ((unsigned)wi >= (unsigned)W0) continue;
                        int tok = ((ti * H0 + hi) * W0 + wi) + 1;
                        const __bf16* ip = in + (long)tok * HD;
                        int widx = kt * 9 + kh * 3 + kw;
#pragma unroll
                        for (int j = 0; j < 3; ++j) {
                            int d = lane + 32 * j;
                            acc[j] += (float)ip[d] * pws[mat][d * 27 + widx];
                        }
                    }
                }
            }
        }
        // LayerNorm over 96 channels (wave32 shuffle reduction)
        float s = acc[0] + acc[1] + acc[2];
#pragma unroll
        for (int m = 16; m >= 1; m >>= 1) s += __shfl_xor(s, m, 32);
        float mean = s * (1.f / 96.f);
        float vs = 0.f;
#pragma unroll
        for (int j = 0; j < 3; ++j) { float t = acc[j] - mean; vs += t * t; }
#pragma unroll
        for (int m = 16; m >= 1; m >>= 1) vs += __shfl_xor(vs, m, 32);
        float inv = rsqrtf(vs * (1.f / 96.f) + 1e-5f);
#pragma unroll
        for (int j = 0; j < 3; ++j) {
            int d = lane + 32 * j;
            float y = (acc[j] - mean) * inv * gs[mat][d] + bs[mat][d];
            if (mat == 0) {
                qpf[((long)bh * NPOOL + o) * HD + d] = y;
                qpb[((long)bh * NPPAD + o) * HD + d] = (__bf16)y;
            } else if (mat == 1) {
                kpb[((long)bh * NPPAD + o) * HD + d] = (__bf16)y;
            } else {
                vtb[((long)bh * HD + d) * NPPAD + o] = (__bf16)y;   // V transposed
            }
        }
    }
}

// ---------------------------------------------------------------------------
// Kernel 5: flash attention per (b,h). One wave per 16-query tile, 32 keys
// per step: S via 6 WMMAs, online softmax (half-wave shuffle row reductions),
// P re-laid out D->A through wave-private LDS, PV via 6 WMMAs. Residual +q
// and scatter to (B*NPOOL, DIM) bf16 rows for the projection GEMM.
// ---------------------------------------------------------------------------
__global__ void __launch_bounds__(256)
k_flash_attn(const __bf16* __restrict__ qpb, const __bf16* __restrict__ kpb,
             const __bf16* __restrict__ vtb, const float* __restrict__ qpf,
             __bf16* __restrict__ aob) {
    __shared__ __align__(16) __bf16 pst[8][16 * 32];
    int wslot = threadIdx.x >> 5;
    int wid = blockIdx.x * (blockDim.x >> 5) + wslot;
    if (wid >= BATCH * NHEADS * QTILES) return;
    int bh = wid / QTILES, qt = wid % QTILES;
    int lane = threadIdx.x & 31, half = lane >> 4, n = lane & 15;

    const __bf16* qb = qpb + ((long)bh * NPPAD + qt * 16) * HD;
    v16bf qf[3];
#pragma unroll
    for (int t = 0; t < 3; ++t) qf[t] = load_fragA16(qb + 32 * t, HD);

    v8f o[6] = {};
    float mrun[8], lrun[8];
#pragma unroll
    for (int r = 0; r < 8; ++r) { mrun[r] = -3.0e30f; lrun[r] = 0.f; }

    const __bf16* kb0 = kpb + (long)bh * NPPAD * HD;
    const __bf16* vb0 = vtb + (long)bh * HD * NPPAD;

    for (int kt = 0; kt < KTILES; ++kt) {
        v8f sa = {}, sb = {};
#pragma unroll
        for (int t = 0; t < 3; ++t) {
            v16bf ka = load_fragB16(kb0 + (long)(kt * 32) * HD + 32 * t, HD);
            v16bf kb = load_fragB16(kb0 + (long)(kt * 32 + 16) * HD + 32 * t, HD);
            sa = wmma_bf16(qf[t], ka, sa);
            sb = wmma_bf16(qf[t], kb, sb);
        }
        int keyA = kt * 32 + n, keyB = keyA + 16;
#pragma unroll
        for (int r = 0; r < 8; ++r) {
            float va = sa[r] * SCALE_F;
            float vb = sb[r] * SCALE_F;
            if (keyA >= NPOOL) va = -3.0e30f;
            if (keyB >= NPOOL) vb = -3.0e30f;
            float tm = fmaxf(va, vb);
#pragma unroll
            for (int m = 8; m >= 1; m >>= 1) tm = fmaxf(tm, __shfl_xor(tm, m, 32));
            float mnew = fmaxf(mrun[r], tm);
            float alpha = __expf(mrun[r] - mnew);
            mrun[r] = mnew;
            float pa = __expf(va - mnew); if (keyA >= NPOOL) pa = 0.f;
            float pb = __expf(vb - mnew); if (keyB >= NPOOL) pb = 0.f;
            float rs = pa + pb;
#pragma unroll
            for (int m = 8; m >= 1; m >>= 1) rs += __shfl_xor(rs, m, 32);
            lrun[r] = lrun[r] * alpha + rs;
#pragma unroll
            for (int f = 0; f < 6; ++f) o[f][r] *= alpha;
            // stage P (16x32 row-major bf16) for A-fragment reload
            pst[wslot][(r + 8 * half) * 32 + n]      = (__bf16)pa;
            pst[wslot][(r + 8 * half) * 32 + 16 + n] = (__bf16)pb;
        }
        __builtin_amdgcn_wave_barrier();
        asm volatile("s_wait_dscnt 0" ::: "memory");
        __builtin_amdgcn_wave_barrier();
        v16bf pf = load_fragA16(&pst[wslot][0], 32);
#pragma unroll
        for (int f = 0; f < 6; ++f) {
            v16bf vf = load_fragB16(vb0 + (long)(f * 16) * NPPAD + kt * 32, NPPAD);
            o[f] = wmma_bf16(pf, vf, o[f]);
        }
        __builtin_amdgcn_wave_barrier();   // LDS reads done before next writes
    }

    int b = bh >> 3, h = bh & 7;
#pragma unroll
    for (int r = 0; r < 8; ++r) {
        int tok = qt * 16 + r + 8 * half;
        if (tok >= NPOOL) continue;
        float rinv = 1.f / lrun[r];
        long qrow = ((long)bh * NPOOL + tok) * HD;
        long orow = ((long)(b * NPOOL + tok)) * DIMC + h * HD;
#pragma unroll
        for (int f = 0; f < 6; ++f) {
            int d = f * 16 + n;
            float val = o[f][r] * rinv + qpf[qrow + d];
            aob[orow + d] = (__bf16)val;
        }
    }
}

// ---------------------------------------------------------------------------
// Kernel 6: output projection GEMM (bf16 WMMA) + bias -> f32 d_out
// ---------------------------------------------------------------------------
__global__ void __launch_bounds__(256)
k_proj_gemm(const __bf16* __restrict__ aob, const __bf16* __restrict__ wpb,
            const float* __restrict__ pbias, float* __restrict__ out) {
    const int NT = DIMC / 16;
    int wave = blockIdx.x * (blockDim.x >> 5) + (threadIdx.x >> 5);
    if (wave >= M2TILES * NT) return;
    int mt = wave / NT, nt = wave % NT;

    v8f acc = {};
    const __bf16* abase = aob + (long)mt * 16 * DIMC;
    const __bf16* bbase = wpb + (long)nt * 16 * DIMC;
#pragma unroll 4
    for (int kk = 0; kk < DIMC; kk += 32) {
        v16bf a = load_fragA16(abase + kk, DIMC);
        v16bf b = load_fragB16(bbase + kk, DIMC);
        acc = wmma_bf16(a, b, acc);
    }
    int lane = threadIdx.x & 31, half = lane >> 4, n = lane & 15;
    int col = nt * 16 + n;
    float bias = pbias[col];
#pragma unroll
    for (int r = 0; r < 8; ++r) {
        int m = mt * 16 + r + 8 * half;
        if (m < M2) out[(long)m * DIMC + col] = acc[r] + bias;
    }
}

// ---------------------------------------------------------------------------
// Host launcher
// ---------------------------------------------------------------------------
extern "C" void kernel_launch(void* const* d_in, const int* in_sizes, int n_in,
                              void* d_out, int out_size, void* d_ws, size_t ws_size,
                              hipStream_t stream) {
    const float* x    = (const float*)d_in[0];
    const float* wq   = (const float*)d_in[1];
    const float* bq   = (const float*)d_in[2];
    const float* wk   = (const float*)d_in[3];
    const float* bk   = (const float*)d_in[4];
    const float* wv   = (const float*)d_in[5];
    const float* bv   = (const float*)d_in[6];
    const float* pwq  = (const float*)d_in[7];
    const float* pwk  = (const float*)d_in[8];
    const float* pwv  = (const float*)d_in[9];
    const float* lgq  = (const float*)d_in[10];
    const float* lbq  = (const float*)d_in[11];
    const float* lgk  = (const float*)d_in[12];
    const float* lbk  = (const float*)d_in[13];
    const float* lgv  = (const float*)d_in[14];
    const float* lbv  = (const float*)d_in[15];
    const float* pw   = (const float*)d_in[16];
    const float* pb   = (const float*)d_in[17];

    // --- workspace carving (256B aligned) ---
    char* ws = (char*)d_ws;
    size_t off = 0;
    auto carve = [&](size_t bytes) -> char* {
        char* p = ws + off;
        off = (off + bytes + 255) & ~(size_t)255;
        return p;
    };
    __bf16* xb   = (__bf16*)carve((size_t)M1PAD * DIMC * 2);           // x bf16, padded rows
    __bf16* wb   = (__bf16*)carve((size_t)4 * DIMC * DIMC * 2);        // wq,wk,wv,proj_w bf16
    __bf16* qpre = (__bf16*)carve((size_t)BATCH * NHEADS * NTOK * HD * 2);
    __bf16* kpre = (__bf16*)carve((size_t)BATCH * NHEADS * NTOK * HD * 2);
    __bf16* vpre = (__bf16*)carve((size_t)BATCH * NHEADS * NTOK * HD * 2);
    float*  qpf  = (float*) carve((size_t)BATCH * NHEADS * NPOOL * HD * 4);
    __bf16* qpb  = (__bf16*)carve((size_t)BATCH * NHEADS * NPPAD * HD * 2);
    __bf16* kpb  = (__bf16*)carve((size_t)BATCH * NHEADS * NPPAD * HD * 2);
    __bf16* vtb  = (__bf16*)carve((size_t)BATCH * NHEADS * HD * NPPAD * 2);
    __bf16* aob  = (__bf16*)carve((size_t)M2PAD * DIMC * 2);

    auto zero = [&](void* p, size_t bytes) {
        long n = (long)(bytes / 4);
        long g = (n + 255) / 256;
        if (g > 4096) g = 4096;
        k_zero_u32<<<(int)g, 256, 0, stream>>>((uint32_t*)p, n);
    };
    // zero padded regions (whole buffers for simplicity; valid parts overwritten)
    zero(xb + (size_t)M1 * DIMC, (size_t)(M1PAD - M1) * DIMC * 2);  // x pad rows
    zero(qpb, (size_t)BATCH * NHEADS * NPPAD * HD * 2);
    zero(kpb, (size_t)BATCH * NHEADS * NPPAD * HD * 2);
    zero(vtb, (size_t)BATCH * NHEADS * HD * NPPAD * 2);
    zero(aob, (size_t)M2PAD * DIMC * 2);

    // convert f32 -> bf16
    k_convert<<<2048, 256, 0, stream>>>(x, wq, wk, wv, pw, xb, wb);

    // QKV GEMM: 1569*48 tiles, 8 waves/block
    {
        int waves = M1TILES * (DIMC / 16);
        k_qkv_gemm<<<(waves + 7) / 8, 256, 0, stream>>>(xb, wb, bq, bk, bv,
                                                        qpre, kpre, vpre);
    }
    // pool + layernorm: 32*1569 waves, 4 waves/block
    {
        int waves = BATCH * NHEADS * NPOOL;
        k_pool_ln<<<(waves + 3) / 4, 128, 0, stream>>>(qpre, kpre, vpre,
                                                       pwq, pwk, pwv,
                                                       lgq, lbq, lgk, lbk, lgv, lbv,
                                                       qpf, qpb, kpb, vtb);
    }
    // flash attention: 32*99 waves, 8 waves/block
    {
        int waves = BATCH * NHEADS * QTILES;
        k_flash_attn<<<(waves + 7) / 8, 256, 0, stream>>>(qpb, kpb, vtb, qpf, aob);
    }
    // output projection: 393*48 tiles
    {
        int waves = M2TILES * (DIMC / 16);
        k_proj_gemm<<<(waves + 7) / 8, 256, 0, stream>>>(aob, wb + (size_t)3 * DIMC * DIMC,
                                                         pb, (float*)d_out);
    }
}
